// TinyCrossAttn_86096914416414
// MI455X (gfx1250) — matrix operand order repaired and checked by reading, not compiled, not run on hardware
//
#include <hip/hip_runtime.h>
#include <stdint.h>

// ---------------------------------------------------------------------------
// TinyCrossAttn for MI455X (gfx1250), compile-only tuned.
//
// Algebraic refactor (exact reassociation):
//   q~ = q_vec @ (wq^T @ wk) * (1/sqrt(D))          [M,D]
//   logits[m,n] = q~[m] . kv[m,n]                   [M,N]
//   attn = softmax_n(logits)                        [M,N]  (output #2)
//   c~[m] = attn[m] @ kv[m]                         [M,D]
//   out  = c~ @ (wo @ wv)^T                         [M,D]  (output #1)
// FLOPs drop 275 GF -> ~2.2 GF; kernel is HBM-bound on ONE pass of kv
// (1.07 GB -> ~46us @ 23.3 TB/s). kv[m] (512x256) is cached bf16 in LDS
// (260KB of the 320KB/WGP) so softmax + attn@kv reuse it with no 2nd pass.
// Dense [2048,256]x[256,256] GEMMs use v_wmma_f32_16x16x32_bf16.
// ---------------------------------------------------------------------------

typedef __attribute__((ext_vector_type(16))) __bf16 v16bf;
typedef __attribute__((ext_vector_type(8)))  float  v8f;

#define M_DIM 2048
#define N_DIM 512
#define D_DIM 256

static __device__ __forceinline__ uint32_t pack_bf16x2(float a, float b) {
    uint32_t ua = __builtin_bit_cast(uint32_t, a);
    uint32_t ub = __builtin_bit_cast(uint32_t, b);
    ua = (ua + 0x7fffu + ((ua >> 16) & 1u)) >> 16;   // RNE
    ub = (ub + 0x7fffu + ((ub >> 16) & 1u)) >> 16;
    return ua | (ub << 16);
}
static __device__ __forceinline__ float bf16lo(uint32_t u) {
    return __builtin_bit_cast(float, u << 16);
}
static __device__ __forceinline__ float bf16hi(uint32_t u) {
    return __builtin_bit_cast(float, u & 0xffff0000u);
}
static __device__ __forceinline__ uint16_t f2bf(float a) {
    uint32_t u = __builtin_bit_cast(uint32_t, a);
    return (uint16_t)((u + 0x7fffu + ((u >> 16) & 1u)) >> 16);
}

// ---------------------------------------------------------------------------
// Weight combine: Bqk[o][k] = sum_e wk[e][o] * wq[e][k]   (= (wq^T wk)^T)
//                 Bov[o][k] = sum_e wo[o][e] * wv[e][k]   (=  wo @ wv  )
// Stored bf16 row-major [out][in] == WMMA B-fragment-friendly layout.
// 16.7M MACs each: trivial.
// ---------------------------------------------------------------------------
__global__ __launch_bounds__(256) void wcombine_qk(const float* __restrict__ wq,
                                                   const float* __restrict__ wk,
                                                   uint16_t* __restrict__ Bqk) {
    const int o = blockIdx.x, k = threadIdx.x;
    float acc = 0.f;
    for (int e = 0; e < D_DIM; ++e)
        acc = fmaf(wk[e * D_DIM + o], wq[e * D_DIM + k], acc);
    Bqk[o * D_DIM + k] = f2bf(acc);
}

__global__ __launch_bounds__(256) void wcombine_ov(const float* __restrict__ wo,
                                                   const float* __restrict__ wv,
                                                   uint16_t* __restrict__ Bov) {
    const int o = blockIdx.x, k = threadIdx.x;
    float acc = 0.f;
    for (int e = 0; e < D_DIM; ++e)
        acc = fmaf(wo[o * D_DIM + e], wv[e * D_DIM + k], acc);
    Bov[o * D_DIM + k] = f2bf(acc);
}

// ---------------------------------------------------------------------------
// C[m,n] = scale * sum_k A[m,k] * Bw[n,k]
// A: fp32 row-major [2048,256]; Bw: bf16 row-major [256,256] ([n][k]).
// 1 block = 4 wave32s; wave w owns a 16x16 output tile, 8 WMMA k-steps.
// ISA fragment layouts (CDNA5 ISA §7.12.2), wave32:
//   A 16x32 bf16 : lane l -> row m=l%16 ; elems 0..7 -> K=kh+i, 8..15 -> K=16+kh+i-8
//                  with kh = (l/16)*8
//   B 32x16 bf16 : lane l -> col n=l%16 ; here elems i -> K=(l/16)*16 + i (contiguous, A loaded likewise)
//   C 16x16 f32  : VGPR r, lane l -> M = r + (l/16)*8, N = l%16
// ---------------------------------------------------------------------------
__global__ __launch_bounds__(128)
void gemm_bf16_xBt(const float* __restrict__ A, const uint16_t* __restrict__ Bw,
                   float* __restrict__ C, float scale) {
    const int lane = threadIdx.x & 31;
    const int wave = threadIdx.x >> 5;
    const int m0 = blockIdx.x * 16;
    const int n0 = (blockIdx.y * 4 + wave) * 16;
    const int r16 = lane & 15;
    const int hv  = lane >> 4;        // lane half
    const int kh  = hv * 8;           // A k sub-offset
    const int bh  = hv * 16;          // B k sub-offset

    const float*    arow = A  + (size_t)(m0 + r16) * D_DIM;
    const uint16_t* brow = Bw + (size_t)(n0 + r16) * D_DIM;

    v8f acc = {};
#pragma unroll
    for (int kk = 0; kk < D_DIM; kk += 32) {
        const float4 f0 = *(const float4*)(arow + kk + 2 * kh);
        const float4 f1 = *(const float4*)(arow + kk + 2 * kh + 4);
        const float4 f2 = *(const float4*)(arow + kk + 2 * kh + 8);
        const float4 f3 = *(const float4*)(arow + kk + 2 * kh + 12);
        v16bf a;
        a[0]  = (__bf16)f0.x; a[1]  = (__bf16)f0.y; a[2]  = (__bf16)f0.z; a[3]  = (__bf16)f0.w;
        a[4]  = (__bf16)f1.x; a[5]  = (__bf16)f1.y; a[6]  = (__bf16)f1.z; a[7]  = (__bf16)f1.w;
        a[8]  = (__bf16)f2.x; a[9]  = (__bf16)f2.y; a[10] = (__bf16)f2.z; a[11] = (__bf16)f2.w;
        a[12] = (__bf16)f3.x; a[13] = (__bf16)f3.y; a[14] = (__bf16)f3.z; a[15] = (__bf16)f3.w;

        const v16bf b = *(const v16bf*)(brow + kk + bh);   // 16 contiguous bf16, 32B aligned

        acc = __builtin_amdgcn_wmma_f32_16x16x32_bf16(
            /*neg_a=*/false, a, /*neg_b=*/false, b,
            /*c_mod=*/(short)0, acc, /*reuse_a=*/false, /*reuse_b=*/false);
    }

    float* crow = C + (size_t)(m0 + hv * 8) * D_DIM + n0 + r16;
#pragma unroll
    for (int r = 0; r < 8; ++r)
        crow[(size_t)r * D_DIM] = acc[r] * scale;
}

// ---------------------------------------------------------------------------
// Per-m attention core. 1 block (512 thr, 16 waves) per m row.
// kv[m] (512x256 f32, 512KB HBM) -> LDS bf16, padded row stride 130 dwords
// (bank = (2n + j) % 64 -> conflict-free for both access phases).
// ---------------------------------------------------------------------------
#define KV_STRIDE_U32 130   // 128 packed bf16x2 + 2 pad dwords

__global__ __launch_bounds__(512)
void attn_core(const float* __restrict__ kv, const float* __restrict__ qt,
               float* __restrict__ ctx, float* __restrict__ attn_out) {
    extern __shared__ char smem_raw[];
    uint32_t* kvb  = (uint32_t*)smem_raw;                       // 512*130 u32
    float*    qrow = (float*)(kvb + N_DIM * KV_STRIDE_U32);     // 256
    float*    lg   = qrow + D_DIM;                              // 512
    float*    red  = lg + N_DIM;                                // 512

    const int m = blockIdx.x;
    const int t = threadIdx.x;

    if (t < D_DIM) qrow[t] = qt[(size_t)m * D_DIM + t];

    // Stream kv[m] once from HBM, store bf16-packed into LDS.
    const float4* src = (const float4*)(kv + (size_t)m * N_DIM * D_DIM);
#pragma unroll 1
    for (int it = 0; it < (N_DIM * D_DIM / 4) / 512; ++it) {
        const int f  = it * 512 + t;          // float4 linear index
        const int n  = f >> 6;                // 64 float4 per row
        const int dc = f & 63;
        const float4 v = src[f];
        uint32_t* dst = kvb + n * KV_STRIDE_U32 + dc * 2;
        dst[0] = pack_bf16x2(v.x, v.y);
        dst[1] = pack_bf16x2(v.z, v.w);
    }
    __syncthreads();

    // logits[n] = q~[m] . kv[m,n]   (scale 1/sqrt(D) pre-folded into q~)
    {
        float acc = 0.f;
        const uint32_t* row = kvb + t * KV_STRIDE_U32;
#pragma unroll 4
        for (int j = 0; j < D_DIM / 2; ++j) {
            const uint32_t u = row[j];
            acc = fmaf(qrow[2 * j],     bf16lo(u), acc);
            acc = fmaf(qrow[2 * j + 1], bf16hi(u), acc);
        }
        lg[t]  = acc;
        red[t] = acc;
    }
    __syncthreads();

    // max-reduce
    for (int s = 256; s > 0; s >>= 1) {
        if (t < s) red[t] = fmaxf(red[t], red[t + s]);
        __syncthreads();
    }
    const float rmax = red[0];
    __syncthreads();

    // exp + sum-reduce
    const float e = __expf(lg[t] - rmax);
    red[t] = e;
    __syncthreads();
    for (int s = 256; s > 0; s >>= 1) {
        if (t < s) red[t] += red[t + s];
        __syncthreads();
    }
    const float inv = 1.0f / red[0];
    const float p   = e * inv;
    lg[t] = p;
    attn_out[(size_t)m * N_DIM + t] = p;      // output #2 (coalesced)
    __syncthreads();

    // c~[m,d] = sum_n p[n] * kv[m,n,d]  (two n-segments of 256 per d)
    {
        const int d   = t & (D_DIM - 1);
        const int seg = t >> 8;
        const int dp  = d >> 1;
        const bool hi = d & 1;
        float acc = 0.f;
        const int nbase = seg * 256;
#pragma unroll 4
        for (int nn = 0; nn < 256; ++nn) {
            const int n = nbase + nn;
            const uint32_t u = kvb[n * KV_STRIDE_U32 + dp];
            acc = fmaf(lg[n], hi ? bf16hi(u) : bf16lo(u), acc);
        }
        __syncthreads();                      // red[0..511] free now
        if (seg == 1) red[d] = acc;
        __syncthreads();
        if (seg == 0) ctx[(size_t)m * D_DIM + d] = acc + red[d];
    }
}

// ---------------------------------------------------------------------------
extern "C" void kernel_launch(void* const* d_in, const int* in_sizes, int n_in,
                              void* d_out, int out_size, void* d_ws, size_t ws_size,
                              hipStream_t stream) {
    (void)in_sizes; (void)n_in; (void)out_size; (void)ws_size;

    const float* q_vec = (const float*)d_in[0];
    const float* kv    = (const float*)d_in[1];
    const float* wq    = (const float*)d_in[2];
    const float* wk    = (const float*)d_in[3];
    const float* wv    = (const float*)d_in[4];
    const float* wo    = (const float*)d_in[5];

    float* out  = (float*)d_out;                              // [2048,256]
    float* attn = (float*)d_out + (size_t)M_DIM * D_DIM;      // [2048,512]

    // Workspace layout (needs ~4.25 MB)
    char* ws = (char*)d_ws;
    uint16_t* Bqk = (uint16_t*)ws;                                  // 128 KB bf16
    uint16_t* Bov = (uint16_t*)(ws + 128 * 1024);                   // 128 KB bf16
    float*    qt  = (float*)(ws + 256 * 1024);                      // 2 MB f32
    float*    ctx = (float*)(ws + 256 * 1024 + 2 * 1024 * 1024);    // 2 MB f32

    // 1) fold weights: (wq^T wk) and (wo wv), bf16 B-ready layout
    wcombine_qk<<<D_DIM, D_DIM, 0, stream>>>(wq, wk, Bqk);
    wcombine_ov<<<D_DIM, D_DIM, 0, stream>>>(wo, wv, Bov);

    // 2) q~ = q_vec @ Bqk^T * (1/sqrt(256))      (WMMA bf16)
    dim3 ggrid(M_DIM / 16, 4);
    gemm_bf16_xBt<<<ggrid, 128, 0, stream>>>(q_vec, Bqk, qt, 0.0625f);

    // 3) fused logits -> softmax -> attn@kv, single HBM pass over kv
    const size_t lds_bytes = (size_t)N_DIM * KV_STRIDE_U32 * 4   // kv bf16 cache
                           + D_DIM * 4 + N_DIM * 4 + N_DIM * 4;  // qrow, lg, red
    attn_core<<<M_DIM, 512, lds_bytes, stream>>>(kv, qt, ctx, attn);

    // 4) out = c~ @ Bov^T                        (WMMA bf16)
    gemm_bf16_xBt<<<ggrid, 128, 0, stream>>>(ctx, Bov, out, 1.0f);
}
